// OneSupremeMemory_50096498540901
// MI455X (gfx1250) — matrix operand order repaired and checked by reading, not compiled, not run on hardware
//
#include <hip/hip_runtime.h>
#include <math.h>

// Problem dimensions (fixed by setup_inputs).
#define BB 8
#define SS 512
#define MM 64
#define TT 576            // S + M
#define HH 768
#define FFD 3072
#define EE 512
#define NLAYER 2
#define NHEAD 12
#define DH 64
#define RR (BB*TT)        // 4608 rows, multiple of 128

typedef __attribute__((ext_vector_type(16))) _Float16 v16h;
typedef __attribute__((ext_vector_type(8)))  float    v8f;

union Frag16 { uint4 q[2]; v16h v; };

// ---------------------------------------------------------------------------
// WMMA GEMM: C[R,N] = A[R,K](f16) @ W[K,N] + bias[N], weights pre-transposed
// to WT[N][K].
// Block: 256 thr (8 waves), tile 128x64, K-stage 64 (2 WMMA K-steps),
// double-buffered LDS.  Tiles are staged with CDNA5 async DMA copies
// (global_load_async_to_lds_b128, ASYNCcnt) issued before the WMMA section,
// so there are no staging VGPRs (round 2/3 builds spilled those to scratch).
// Sync per stage: s_wait_asynccnt 0 + workgroup barrier.
// Note: the instruction's inst_offset is added to BOTH the LDS address and
// the global address (ISA 08_async_tensor 4.4), so one LDS-offset VGPR and
// one global-address VGPR pair cover all chunks via offset:0/16/32/48.
// Wave w: rows (w>>1)*32, cols (w&1)*32 -> 2x2 grid of 16x16 WMMA tiles.
// A fragment (16x32 f16, ISA 7.12.2): lane L holds row L&15;
//   lanes 0-15: elems 0-7 = K 0..7, elems 8-15 = K 16..23
//   lanes 16-31: elems 0-7 = K 8..15, elems 8-15 = K 24..31
// B fragment (32x16): lane L holds col L&15; elems in the same K order as A.
// C/D (16x16 f32): vgpr j: lanes 0-15 -> M=j, N=lane; lanes 16-31 -> M=8+j.
// ---------------------------------------------------------------------------
#define GTM 128
#define GTN 64
#define GTK2 64
#define LPAD 8
#define LDK (GTK2 + LPAD)    // 72 halves = 144B rows (multiple of 16B)

__global__ __launch_bounds__(256, 1) void osm_gemm_f16(
    const _Float16* __restrict__ A, const _Float16* __restrict__ WT,
    const float* __restrict__ bias, float* __restrict__ C, int K, int N)
{
  __shared__ __align__(16) _Float16 As[2][GTM][LDK];
  __shared__ __align__(16) _Float16 Bs[2][GTN][LDK];
  const int tid  = threadIdx.x;
  const int lane = tid & 31;
  const int wave = tid >> 5;
  const int rowBase = blockIdx.y * GTM;
  const int colBase = blockIdx.x * GTN;
  const int wm = (wave >> 1) * 32;
  const int wn = (wave & 1) * 32;

  const int arow = tid >> 1;          // 0..127
  const int akc  = (tid & 1) * 32;    // 0 or 32
  const int bn   = tid >> 2;          // 0..63
  const int bkc  = (tid & 3) * 16;    // 0,16,32,48

  const _Float16* ga = A  + (size_t)(rowBase + arow) * K + akc;
  const _Float16* gb = WT + (size_t)(colBase + bn)   * K + bkc;

  // Wave-relative LDS byte offsets of this thread's store slots (flat LDS
  // aperture pointers carry the offset in the low 32 bits).
  const unsigned ldsA0 = (unsigned)(unsigned long long)&As[0][arow][akc];
  const unsigned ldsA1 = (unsigned)(unsigned long long)&As[1][arow][akc];
  const unsigned ldsB0 = (unsigned)(unsigned long long)&Bs[0][bn][bkc];
  const unsigned ldsB1 = (unsigned)(unsigned long long)&Bs[1][bn][bkc];

  auto stage = [&](int buf, int k) {
    const _Float16* pa = ga + k;
    const _Float16* pb = gb + k;
    const unsigned la = buf ? ldsA1 : ldsA0;
    const unsigned lb = buf ? ldsB1 : ldsB0;
    asm volatile("global_load_async_to_lds_b128 %0, %1, off"           :: "v"(la), "v"(pa) : "memory");
    asm volatile("global_load_async_to_lds_b128 %0, %1, off offset:16" :: "v"(la), "v"(pa) : "memory");
    asm volatile("global_load_async_to_lds_b128 %0, %1, off offset:32" :: "v"(la), "v"(pa) : "memory");
    asm volatile("global_load_async_to_lds_b128 %0, %1, off offset:48" :: "v"(la), "v"(pa) : "memory");
    asm volatile("global_load_async_to_lds_b128 %0, %1, off"           :: "v"(lb), "v"(pb) : "memory");
    asm volatile("global_load_async_to_lds_b128 %0, %1, off offset:16" :: "v"(lb), "v"(pb) : "memory");
  };

  stage(0, 0);
  asm volatile("s_wait_asynccnt 0x0" ::: "memory");
  __syncthreads();

  v8f acc[2][2] = {};
  const int mlane = lane & 15;
  const int kb0 = (lane >> 4) * 8;
  const int ks0 = (lane >> 4) * 16;

  int cur = 0;
  for (int k0 = 0; k0 < K; k0 += GTK2) {
    const bool more = (k0 + GTK2) < K;
    if (more) {
      stage(cur ^ 1, k0 + GTK2);   // async DMA overlaps the WMMA section
      if (k0 + 2*GTK2 < K) {       // CDNA5 global_prefetch_b8, stage after next
        __builtin_prefetch((const void*)(ga + k0 + 2*GTK2), 0, 1);
        __builtin_prefetch((const void*)(gb + k0 + 2*GTK2), 0, 1);
      }
    }
    #pragma unroll
    for (int s = 0; s < 2; ++s) {           // two WMMA K-steps per stage
      Frag16 fa[2], fb[2];
      const int kb = s*32 + kb0;
      const int ks = s*32 + ks0;
      #pragma unroll
      for (int i = 0; i < 2; ++i) {
        const int r = wm + i*16 + mlane;
        fa[i].q[0] = *(const uint4*)&As[cur][r][kb];
        fa[i].q[1] = *(const uint4*)&As[cur][r][kb + 16];
        const int c = wn + i*16 + mlane;
        fb[i].q[0] = *(const uint4*)&Bs[cur][c][kb];
        fb[i].q[1] = *(const uint4*)&Bs[cur][c][kb + 16];
      }
      #pragma unroll
      for (int mi = 0; mi < 2; ++mi)
        #pragma unroll
        for (int ni = 0; ni < 2; ++ni)
          acc[mi][ni] = __builtin_amdgcn_wmma_f32_16x16x32_f16(
              false, fa[mi].v, false, fb[ni].v, (short)0, acc[mi][ni], false, false);
    }
    if (more) {
      asm volatile("s_wait_asynccnt 0x0" ::: "memory");
      __syncthreads();                     // single barrier per stage
      cur ^= 1;
    }
  }

  const int madd = (lane >> 4) * 8;
  #pragma unroll
  for (int mi = 0; mi < 2; ++mi) {
    #pragma unroll
    for (int ni = 0; ni < 2; ++ni) {
      const int colc = colBase + wn + ni * 16 + mlane;
      const float bv = bias ? bias[colc] : 0.0f;
      #pragma unroll
      for (int j = 0; j < 8; ++j) {
        const int rowc = rowBase + wm + mi * 16 + madd + j;
        C[(size_t)rowc * N + colc] = acc[mi][ni][j] + bv;
      }
    }
  }
}

// ---------------------------------------------------------------------------
// f32 W[K][N] -> f16 WT[N][K] tiled transpose (32x32 tiles through LDS).
// ---------------------------------------------------------------------------
__global__ __launch_bounds__(256) void osm_conv_t(
    const float* __restrict__ W, _Float16* __restrict__ WT, int K, int N)
{
  __shared__ float tile[32][33];
  const int tx = threadIdx.x & 31;
  const int ty = threadIdx.x >> 5;   // 0..7
  const int k0 = blockIdx.y * 32;
  const int n0 = blockIdx.x * 32;
  #pragma unroll
  for (int i = 0; i < 4; ++i) {
    const int k = ty + 8*i;
    tile[k][tx] = W[(size_t)(k0 + k) * N + (n0 + tx)];
  }
  __syncthreads();
  #pragma unroll
  for (int i = 0; i < 4; ++i) {
    const int n = ty + 8*i;
    WT[(size_t)(n0 + n) * K + (k0 + tx)] = (_Float16)tile[tx][n];
  }
}

// ---------------------------------------------------------------------------
// Stable partition of indices by attention mask (argsort(1-m) stable).
// ---------------------------------------------------------------------------
__global__ void osm_build_order(const int* __restrict__ mask,
                                int* __restrict__ order, int* __restrict__ Lbuf)
{
  const int b = blockIdx.x;
  if (threadIdx.x == 0) {
    int c = 0;
    for (int i = 0; i < SS; ++i) if (mask[b*SS+i] != 0) order[b*SS + c++] = i;
    Lbuf[b] = c;
    for (int i = 0; i < SS; ++i) if (mask[b*SS+i] == 0) order[b*SS + c++] = i;
  }
}

// ---------------------------------------------------------------------------
// Gather tokens/memory, add pos+type embeddings, LayerNorm; emit f32 + f16,
// attention bias row and token-type buffer.
// ---------------------------------------------------------------------------
__global__ __launch_bounds__(256) void osm_embed_ln(
    const int* __restrict__ ids, const int* __restrict__ tty,
    const int* __restrict__ order, const int* __restrict__ Lbuf,
    const float* __restrict__ word_emb, const float* __restrict__ pos_emb,
    const float* __restrict__ type_emb, const float* __restrict__ memory,
    const float* __restrict__ gs, const float* __restrict__ gb,
    float* __restrict__ X, _Float16* __restrict__ Xh,
    float* __restrict__ battn, int* __restrict__ ttbuf)
{
  const int p = blockIdx.x, b = blockIdx.y, tid = threadIdx.x;
  const int Lb = Lbuf[b];
  const float* rowp; int ttv;
  if (p < Lb) {
    const int si = order[b*SS + p];
    rowp = word_emb + (size_t)ids[b*SS + si] * HH;
    int t0 = tty[b*SS + si]; ttv = t0 < 0 ? 0 : (t0 > 1 ? 1 : t0);
  } else if (p < Lb + MM) {
    rowp = memory + (size_t)(p - Lb) * HH; ttv = 1;
  } else {
    int sp = p - MM; sp = sp < 0 ? 0 : (sp > SS-1 ? SS-1 : sp);
    const int si = order[b*SS + sp];
    rowp = word_emb + (size_t)ids[b*SS + si] * HH; ttv = 1;
  }
  float v[3]; float s = 0.f, s2 = 0.f;
  #pragma unroll
  for (int i = 0; i < 3; ++i) {
    const int d = tid + 256*i;
    const float e = rowp[d] + pos_emb[p*HH + d] + type_emb[ttv*HH + d];
    v[i] = e; s += e; s2 += e*e;
  }
  __shared__ float r1[256], r2[256];
  r1[tid] = s; r2[tid] = s2; __syncthreads();
  for (int o = 128; o > 0; o >>= 1) {
    if (tid < o) { r1[tid] += r1[tid+o]; r2[tid] += r2[tid+o]; }
    __syncthreads();
  }
  const float mean = r1[0] * (1.0f/HH);
  const float var  = fmaxf(r2[0] * (1.0f/HH) - mean*mean, 0.f);
  const float rstd = rsqrtf(var + 1e-12f);
  const size_t roff = (size_t)(b*TT + p) * HH;
  #pragma unroll
  for (int i = 0; i < 3; ++i) {
    const int d = tid + 256*i;
    const float o = (v[i]-mean)*rstd*gs[d] + gb[d];
    X[roff + d] = o; Xh[roff + d] = (_Float16)o;
  }
  if (tid == 0) {
    battn[b*TT + p] = (p < Lb + MM) ? 0.0f : -10000.0f;
    ttbuf[b*TT + p] = ttv;
  }
}

// Residual add + LayerNorm (in place on X), emits f16 copy.
__global__ __launch_bounds__(256) void osm_add_ln(
    float* __restrict__ X, const float* __restrict__ Y,
    const float* __restrict__ gs, const float* __restrict__ gb,
    _Float16* __restrict__ Xh)
{
  const int row = blockIdx.x, tid = threadIdx.x;
  __shared__ float r1[256], r2[256];
  float v[3]; float s = 0.f, s2 = 0.f;
  const size_t roff = (size_t)row * HH;
  #pragma unroll
  for (int i = 0; i < 3; ++i) {
    const int d = tid + 256*i;
    const float e = X[roff + d] + Y[roff + d];
    v[i] = e; s += e; s2 += e*e;
  }
  r1[tid] = s; r2[tid] = s2; __syncthreads();
  for (int o = 128; o > 0; o >>= 1) {
    if (tid < o) { r1[tid] += r1[tid+o]; r2[tid] += r2[tid+o]; }
    __syncthreads();
  }
  const float mean = r1[0] * (1.0f/HH);
  const float var  = fmaxf(r2[0] * (1.0f/HH) - mean*mean, 0.f);
  const float rstd = rsqrtf(var + 1e-12f);
  #pragma unroll
  for (int i = 0; i < 3; ++i) {
    const int d = tid + 256*i;
    const float o = (v[i]-mean)*rstd*gs[d] + gb[d];
    X[roff + d] = o; Xh[roff + d] = (_Float16)o;
  }
}

// ---------------------------------------------------------------------------
// Attention: one block per (q, b*NH). 64 threads. softmax(QK^T/8 + bias) @ V.
// ---------------------------------------------------------------------------
__global__ __launch_bounds__(64) void osm_attn(
    const float* __restrict__ Q, const float* __restrict__ Kk,
    const float* __restrict__ V, const float* __restrict__ battn,
    float* __restrict__ Ctx)
{
  const int q  = blockIdx.x;
  const int bh = blockIdx.y;
  const int b = bh / NHEAD, h = bh % NHEAD;
  const int tid = threadIdx.x;
  __shared__ float sbuf[TT];
  __shared__ float red[64];
  __shared__ float qs[DH];
  const size_t base = (size_t)(b*TT) * HH + h*DH;
  qs[tid] = Q[base + (size_t)q*HH + tid];
  __syncthreads();
  float lmax = -3.0e38f;
  for (int kk = tid; kk < TT; kk += 64) {
    const float* kp = Kk + base + (size_t)kk * HH;
    float s = 0.f;
    #pragma unroll 16
    for (int d = 0; d < DH; ++d) s += qs[d] * kp[d];
    s = s * 0.125f + battn[b*TT + kk];
    sbuf[kk] = s;
    lmax = fmaxf(lmax, s);
  }
  red[tid] = lmax; __syncthreads();
  for (int o = 32; o > 0; o >>= 1) { if (tid < o) red[tid] = fmaxf(red[tid], red[tid+o]); __syncthreads(); }
  const float mx = red[0];
  __syncthreads();
  float lsum = 0.f;
  for (int kk = tid; kk < TT; kk += 64) { const float e = __expf(sbuf[kk]-mx); sbuf[kk] = e; lsum += e; }
  red[tid] = lsum; __syncthreads();
  for (int o = 32; o > 0; o >>= 1) { if (tid < o) red[tid] += red[tid+o]; __syncthreads(); }
  const float inv = 1.0f / red[0];
  __syncthreads();
  float accd = 0.f;
  for (int kk = 0; kk < TT; ++kk) accd += sbuf[kk] * V[base + (size_t)kk*HH + tid];
  Ctx[base + (size_t)q*HH + tid] = accd * inv;
}

// ---------------------------------------------------------------------------
// Elementwise kernels.
// ---------------------------------------------------------------------------
__global__ void osm_f2h(const float* __restrict__ in, _Float16* __restrict__ out, int n) {
  const int i = blockIdx.x*256 + threadIdx.x;
  if (i < n) out[i] = (_Float16)in[i];
}
__global__ void osm_gelu_h(const float* __restrict__ in, _Float16* __restrict__ out, int n) {
  const int i = blockIdx.x*256 + threadIdx.x;
  if (i < n) { const float x = in[i]; out[i] = (_Float16)(0.5f*x*(1.0f + erff(x*0.70710678118f))); }
}
__global__ void osm_relu_h(const float* __restrict__ in, _Float16* __restrict__ out, int n) {
  const int i = blockIdx.x*256 + threadIdx.x;
  if (i < n) out[i] = (_Float16)fmaxf(in[i], 0.0f);
}
__global__ void osm_sigmoid(float* __restrict__ d, int n) {
  const int i = blockIdx.x*256 + threadIdx.x;
  if (i < n) d[i] = 1.0f/(1.0f + __expf(-d[i]));
}
__global__ void osm_tanh(float* __restrict__ d, int n) {
  const int i = blockIdx.x*256 + threadIdx.x;
  if (i < n) d[i] = tanhf(d[i]);
}

// Masked softmax over T, weighted sum, final relu. One thread per (b,e).
__global__ __launch_bounds__(64) void osm_pool(
    const float* __restrict__ Wt, const float* __restrict__ Vt,
    const int* __restrict__ ttb, float* __restrict__ out)
{
  const int b = blockIdx.y;
  const int e = blockIdx.x*64 + threadIdx.x;
  float mx = -3.0e38f;
  for (int t = 0; t < TT; ++t)
    if (ttb[b*TT + t] != 1) mx = fmaxf(mx, Wt[(size_t)(b*TT + t)*EE + e]);
  float se = 0.f, acc = 0.f;
  for (int t = 0; t < TT; ++t) {
    if (ttb[b*TT + t] != 1) {
      const float ev = __expf(Wt[(size_t)(b*TT + t)*EE + e] - mx);
      se += ev;
      acc += ev * Vt[(size_t)(b*TT + t)*EE + e];
    }
  }
  const float r = (se > 0.f) ? (acc/se) : 0.f;
  out[b*EE + e] = fmaxf(r, 0.f);
}

// ---------------------------------------------------------------------------
extern "C" void kernel_launch(void* const* d_in, const int* in_sizes, int n_in,
                              void* d_out, int out_size, void* d_ws, size_t ws_size,
                              hipStream_t stream)
{
  (void)in_sizes; (void)n_in; (void)out_size; (void)ws_size;
  const int*   input_ids      = (const int*)d_in[0];
  const int*   token_type_ids = (const int*)d_in[1];
  const int*   attention_mask = (const int*)d_in[2];
  const float* word_emb = (const float*)d_in[3];
  const float* pos_emb  = (const float*)d_in[4];
  const float* type_emb = (const float*)d_in[5];
  const float* emb_ln_s = (const float*)d_in[6];
  const float* emb_ln_b = (const float*)d_in[7];
  const float* memory   = (const float*)d_in[8];
  const float* q_w = (const float*)d_in[9];   const float* q_b = (const float*)d_in[10];
  const float* k_w = (const float*)d_in[11];  const float* k_b = (const float*)d_in[12];
  const float* v_w = (const float*)d_in[13];  const float* v_b = (const float*)d_in[14];
  const float* o_w = (const float*)d_in[15];  const float* o_b = (const float*)d_in[16];
  const float* ln1_s = (const float*)d_in[17]; const float* ln1_b = (const float*)d_in[18];
  const float* f_w1 = (const float*)d_in[19]; const float* f_b1 = (const float*)d_in[20];
  const float* f_w2 = (const float*)d_in[21]; const float* f_b2 = (const float*)d_in[22];
  const float* ln2_s = (const float*)d_in[23]; const float* ln2_b = (const float*)d_in[24];
  const float* key_w1 = (const float*)d_in[25]; const float* key_b1 = (const float*)d_in[26];
  const float* key_w2 = (const float*)d_in[27]; const float* key_b2 = (const float*)d_in[28];
  const float* val_w1 = (const float*)d_in[29]; const float* val_b1 = (const float*)d_in[30];
  const float* val_w2 = (const float*)d_in[31]; const float* val_b2 = (const float*)d_in[32];

  char* base = (char*)d_ws;
  size_t off = 0;
  auto alloc = [&](size_t bytes) -> void* {
    void* p = base + off;
    off = (off + bytes + 255) & ~(size_t)255;
    return p;
  };

  int*   order = (int*)alloc(sizeof(int)*BB*SS);
  int*   Lbuf  = (int*)alloc(sizeof(int)*BB);
  int*   ttbuf = (int*)alloc(sizeof(int)*BB*TT);
  float* battn = (float*)alloc(sizeof(float)*BB*TT);
  float*    X   = (float*)alloc(sizeof(float)*(size_t)RR*HH);
  _Float16* Xh  = (_Float16*)alloc(2ull*RR*HH);
  float*    Qf  = (float*)alloc(sizeof(float)*(size_t)RR*HH);
  float*    Kf  = (float*)alloc(sizeof(float)*(size_t)RR*HH);
  float*    Vf  = (float*)alloc(sizeof(float)*(size_t)RR*HH);
  float*    Ctx = (float*)alloc(sizeof(float)*(size_t)RR*HH);
  _Float16* Ctxh= (_Float16*)alloc(2ull*RR*HH);
  float*    TmpH= (float*)alloc(sizeof(float)*(size_t)RR*HH);
  float*    Hbuf= (float*)alloc(sizeof(float)*(size_t)RR*FFD);
  _Float16* Hh  = (_Float16*)alloc(2ull*RR*FFD);
  float*    KH  = (float*)alloc(sizeof(float)*(size_t)RR*1536);   // key/val hidden (reused)
  _Float16* KHh = (_Float16*)alloc(2ull*RR*1536);
  float*    Wgt = (float*)alloc(sizeof(float)*(size_t)RR*EE);
  float*    Val = (float*)alloc(sizeof(float)*(size_t)RR*EE);
  // f16 transposed-weight arena (all stored as WT[N][K])
  _Float16* qwh  = (_Float16*)alloc(2ull*NLAYER*HH*HH);
  _Float16* kwh  = (_Float16*)alloc(2ull*NLAYER*HH*HH);
  _Float16* vwh  = (_Float16*)alloc(2ull*NLAYER*HH*HH);
  _Float16* owh  = (_Float16*)alloc(2ull*NLAYER*HH*HH);
  _Float16* f1h  = (_Float16*)alloc(2ull*NLAYER*HH*FFD);
  _Float16* f2h_ = (_Float16*)alloc(2ull*NLAYER*FFD*HH);
  _Float16* kw1h = (_Float16*)alloc(2ull*HH*1536);
  _Float16* kw2h = (_Float16*)alloc(2ull*1536*EE);
  _Float16* vw1h = (_Float16*)alloc(2ull*HH*1024);
  _Float16* vw2h = (_Float16*)alloc(2ull*1024*EE);

  auto eg = [](int n) { return dim3((unsigned)((n + 255) / 256)); };
  auto convT = [&](const float* src, _Float16* dst, int K, int N) {
    osm_conv_t<<<dim3((unsigned)(N/32), (unsigned)(K/32)), 256, 0, stream>>>(src, dst, K, N);
  };
  auto gemm = [&](const _Float16* A, const _Float16* WT, const float* bias,
                  float* C, int Rr, int Kk, int Nn) {
    dim3 g((unsigned)(Nn / GTN), (unsigned)(Rr / GTM));
    osm_gemm_f16<<<g, 256, 0, stream>>>(A, WT, bias, C, Kk, Nn);
  };

  // Preprocessing: mask partition + weight transpose/convert (per layer slice)
  osm_build_order<<<BB, 32, 0, stream>>>(attention_mask, order, Lbuf);
  for (int l = 0; l < NLAYER; ++l) {
    convT(q_w + (size_t)l*HH*HH, qwh + (size_t)l*HH*HH, HH, HH);
    convT(k_w + (size_t)l*HH*HH, kwh + (size_t)l*HH*HH, HH, HH);
    convT(v_w + (size_t)l*HH*HH, vwh + (size_t)l*HH*HH, HH, HH);
    convT(o_w + (size_t)l*HH*HH, owh + (size_t)l*HH*HH, HH, HH);
    convT(f_w1 + (size_t)l*HH*FFD, f1h + (size_t)l*HH*FFD, HH, FFD);
    convT(f_w2 + (size_t)l*FFD*HH, f2h_ + (size_t)l*FFD*HH, FFD, HH);
  }
  convT(key_w1, kw1h, HH, 1536);
  convT(key_w2, kw2h, 1536, EE);
  convT(val_w1, vw1h, HH, 1024);
  convT(val_w2, vw2h, 1024, EE);

  osm_embed_ln<<<dim3(TT, BB), 256, 0, stream>>>(
      input_ids, token_type_ids, order, Lbuf, word_emb, pos_emb, type_emb,
      memory, emb_ln_s, emb_ln_b, X, Xh, battn, ttbuf);

  // Transformer layers
  for (int l = 0; l < NLAYER; ++l) {
    const size_t wo = (size_t)l*HH*HH;
    gemm(Xh, qwh + wo, q_b + l*HH, Qf, RR, HH, HH);
    gemm(Xh, kwh + wo, k_b + l*HH, Kf, RR, HH, HH);
    gemm(Xh, vwh + wo, v_b + l*HH, Vf, RR, HH, HH);
    osm_attn<<<dim3(TT, BB*NHEAD), 64, 0, stream>>>(Qf, Kf, Vf, battn, Ctx);
    osm_f2h<<<eg(RR*HH), 256, 0, stream>>>(Ctx, Ctxh, RR*HH);
    gemm(Ctxh, owh + wo, o_b + l*HH, TmpH, RR, HH, HH);
    osm_add_ln<<<RR, 256, 0, stream>>>(X, TmpH, ln1_s + l*HH, ln1_b + l*HH, Xh);
    gemm(Xh, f1h + (size_t)l*HH*FFD, f_b1 + l*FFD, Hbuf, RR, HH, FFD);
    osm_gelu_h<<<eg(RR*FFD), 256, 0, stream>>>(Hbuf, Hh, RR*FFD);
    gemm(Hh, f2h_ + (size_t)l*FFD*HH, f_b2 + l*HH, TmpH, RR, FFD, HH);
    osm_add_ln<<<RR, 256, 0, stream>>>(X, TmpH, ln2_s + l*HH, ln2_b + l*HH, Xh);
  }

  // Pooling head: weight = sigmoid(relu(x@key_w1+b)@key_w2+b)
  gemm(Xh, kw1h, key_b1, KH, RR, HH, 1536);
  osm_relu_h<<<eg(RR*1536), 256, 0, stream>>>(KH, KHh, RR*1536);
  gemm(KHh, kw2h, key_b2, Wgt, RR, 1536, EE);
  osm_sigmoid<<<eg(RR*EE), 256, 0, stream>>>(Wgt, RR*EE);
  // value = tanh(relu(x@val_w1+b)@val_w2+b)
  gemm(Xh, vw1h, val_b1, KH, RR, HH, 1024);
  osm_relu_h<<<eg(RR*1024), 256, 0, stream>>>(KH, KHh, RR*1024);
  gemm(KHh, vw2h, val_b2, Val, RR, 1024, EE);
  osm_tanh<<<eg(RR*EE), 256, 0, stream>>>(Val, RR*EE);

  osm_pool<<<dim3(EE/64, BB), 64, 0, stream>>>(Wgt, Val, ttbuf, (float*)d_out);
}
